// HierarchicalEvidentialGAT_83829171683614
// MI455X (gfx1250) — compile-verified, matrix operand order checked by reading, not run on hardware
//
#include <hip/hip_runtime.h>
#include <hip/hip_bf16.h>
#include <math.h>

// ---------------------------------------------------------------------------
// Types for CDNA5 WMMA
// ---------------------------------------------------------------------------
typedef __attribute__((ext_vector_type(16))) __bf16 v16bf;
typedef __attribute__((ext_vector_type(8)))  __bf16 v8bf;
typedef __attribute__((ext_vector_type(8)))  float  v8f;

union FragBF  { v16bf v; __bf16 e[16]; };
union Pack8BF { v8bf  v; __bf16 e[8];  };

__device__ __forceinline__ __bf16 f2bf(float f) {
    unsigned u = __builtin_bit_cast(unsigned, f);
    unsigned r = (u + 0x7FFFu + ((u >> 16) & 1u)) >> 16;   // round-to-nearest-even
    unsigned short s = (unsigned short)r;
    return __builtin_bit_cast(__bf16, s);
}

// ordered-uint encoding of float for atomicMax
__device__ __forceinline__ unsigned fkey(float f) {
    unsigned u = __builtin_bit_cast(unsigned, f);
    return (u & 0x80000000u) ? ~u : (u | 0x80000000u);
}
__device__ __forceinline__ float fval(unsigned k) {
    unsigned u = (k & 0x80000000u) ? (k & 0x7FFFFFFFu) : ~k;
    return __builtin_bit_cast(float, u);
}

__device__ __forceinline__ float softplus_f(float x) {
    return (x > 20.f) ? x : log1pf(expf(x));
}

// ---------------------------------------------------------------------------
// WMMA GEMM:  C[M,N] = A[M,K] (fp32 row-major, lda) * B[K,N] (fp32 row-major, ldb)
// bf16 inputs, fp32 accumulate via v_wmma_f32_16x16x32_bf16.
// Block = 256 threads (8 waves), tile 64x64x32, DOUBLE-BUFFERED LDS:
// next K-chunk's global loads are issued before the current chunk's WMMAs,
// converted+stored to the alternate LDS buffer, single barrier per iteration.
//
// LDS tiles live in WMMA *fragment order*: [strip/subtile][lane][16 bf16],
// so each lane's fragment is one contiguous 32-byte LDS read and each staging
// thread issues one 16-byte LDS store.
//
// 16-bit A layout (ISA 7.12.2): lanes 0-15 row M=L, K in {0-7,16-23};
// lanes 16-31 row M=L-16, K in {8-15,24-31}.  B (32x16) mirrors with N=L&15.
// Requires: N % 64 == 0, K % 32 == 0.  M guarded.
// ---------------------------------------------------------------------------
#define GBM 64
#define GBN 64
#define GBK 32

__global__ __launch_bounds__(256) void heg_gemm_bf16(
    const float* __restrict__ A, const float* __restrict__ B,
    float* __restrict__ C, const float* __restrict__ bias,
    int M, int N, int K, int lda, int ldb, int ldc, int relu)
{
    __shared__ __bf16 AsL[2][4 * 32 * 16] __attribute__((aligned(32)));
    __shared__ __bf16 BsL[2][4 * 32 * 16] __attribute__((aligned(32)));

    const int tid  = threadIdx.x;
    const int lane = tid & 31;
    const int wave = tid >> 5;        // 0..7
    const int wm   = wave & 3;        // 16-row strip of the 64-row tile
    const int wn   = wave >> 2;       // 32-col half of the 64-col tile

    const int row0 = blockIdx.y * GBM;
    const int col0 = blockIdx.x * GBN;

    // ---- staging assignment: A ----
    const int ar      = tid >> 2;                         // 0..63 (row in tile)
    const int ag      = tid & 3;                          // 0..3  (k-group of 8)
    const int a_lane  = (ar & 15) + ((ag & 1) << 4);      // hi k-groups -> lanes 16-31
    const int a_off   = (ar >> 4) * 512 + a_lane * 16 + ((ag >> 1) << 3);
    const int gr      = row0 + ar;
    const float* Aptr = A + (size_t)gr * lda + (ag << 3);

    // ---- staging assignment: B ----
    const int bkg     = tid >> 6;                         // 0..3 (k-group of 8)
    const int bn      = tid & 63;                         // 0..63 (col in tile)
    const int b_lane  = (bn & 15) + ((bkg & 1) << 4);
    const int b_off   = (bn >> 4) * 512 + b_lane * 16 + ((bkg >> 1) << 3);
    const float* Bptr = B + (size_t)(bkg << 3) * ldb + col0 + bn;

    auto loadA = [&](int kk, float* r) {
        if (gr < M) {
            const float4 f0 = *(const float4*)(Aptr + kk);
            const float4 f1 = *(const float4*)(Aptr + kk + 4);
            r[0] = f0.x; r[1] = f0.y; r[2] = f0.z; r[3] = f0.w;
            r[4] = f1.x; r[5] = f1.y; r[6] = f1.z; r[7] = f1.w;
        } else {
            #pragma unroll
            for (int j = 0; j < 8; ++j) r[j] = 0.f;
        }
    };
    auto loadB = [&](int kk, float* r) {
        #pragma unroll
        for (int j = 0; j < 8; ++j) r[j] = Bptr[(size_t)(kk + j) * ldb];
    };
    auto storeA = [&](int buf, const float* r) {
        Pack8BF p8;
        #pragma unroll
        for (int j = 0; j < 8; ++j) p8.e[j] = f2bf(r[j]);
        *(v8bf*)&AsL[buf][a_off] = p8.v;
    };
    auto storeB = [&](int buf, const float* r) {
        Pack8BF p8;
        #pragma unroll
        for (int j = 0; j < 8; ++j) p8.e[j] = f2bf(r[j]);
        *(v8bf*)&BsL[buf][b_off] = p8.v;
    };

    v8f acc0 = {};
    v8f acc1 = {};

    // prologue: stage chunk 0 into buffer 0
    {
        float ra[8], rb[8];
        loadA(0, ra);
        loadB(0, rb);
        storeA(0, ra);
        storeB(0, rb);
    }
    __syncthreads();

    int p = 0;
    for (int k0 = 0; k0 < K; k0 += GBK) {
        const bool more = (k0 + GBK) < K;

        // issue next chunk's global loads before consuming current tile
        float ra[8], rb[8];
        if (more) {
            loadA(k0 + GBK, ra);
            loadB(k0 + GBK, rb);
            if (k0 + 2 * GBK < K) {
                if (gr < M) __builtin_prefetch(Aptr + k0 + 2 * GBK, 0, 1);
                __builtin_prefetch(Bptr + (size_t)(k0 + 2 * GBK) * ldb, 0, 1);
            }
        }

        // fragments: one contiguous 32B LDS read per lane
        FragBF fa, fb0, fb1;
        fa.v  = *(const v16bf*)&AsL[p][wm * 512 + lane * 16];
        fb0.v = *(const v16bf*)&BsL[p][((wn << 1) + 0) * 512 + lane * 16];
        fb1.v = *(const v16bf*)&BsL[p][((wn << 1) + 1) * 512 + lane * 16];

        acc0 = __builtin_amdgcn_wmma_f32_16x16x32_bf16(
            false, fa.v, false, fb0.v, (short)0, acc0, false, false);
        acc1 = __builtin_amdgcn_wmma_f32_16x16x32_bf16(
            false, fa.v, false, fb1.v, (short)0, acc1, false, false);

        if (more) {
            storeA(p ^ 1, ra);
            storeB(p ^ 1, rb);
            __syncthreads();
            p ^= 1;
        }
    }

    // epilogue: C/D layout -> lane L: N = L&15 ; VGPR j: M = j + 8*(L>=16)
    #pragma unroll
    for (int t = 0; t < 2; ++t) {
        v8f acc = t ? acc1 : acc0;
        int n = col0 + (wn << 5) + (t << 4) + (lane & 15);
        float bv = bias ? bias[n] : 0.f;
        #pragma unroll
        for (int j = 0; j < 8; ++j) {
            int m = row0 + (wm << 4) + j + ((lane >> 4) << 3);
            if (m < M) {
                float v = acc[j] + bv;
                if (relu) v = fmaxf(v, 0.f);
                C[(size_t)m * ldc + n] = v;
            }
        }
    }
}

// ---------------------------------------------------------------------------
// s1[i] = Wh[i,:] . a[0:H] ; s2[i] = Wh[i,:] . a[H:2H]   (one wave per row)
// ---------------------------------------------------------------------------
__global__ __launch_bounds__(256) void heg_rowdot2(
    const float* __restrict__ Wh, const float* __restrict__ a,
    float* __restrict__ s1, float* __restrict__ s2, int N, int H)
{
    int row  = blockIdx.x * (blockDim.x >> 5) + (threadIdx.x >> 5);
    int lane = threadIdx.x & 31;
    if (row >= N) return;
    float d1 = 0.f, d2 = 0.f;
    for (int c = lane; c < H; c += 32) {
        float w = Wh[(size_t)row * H + c];
        d1 += w * a[c];
        d2 += w * a[H + c];
    }
    #pragma unroll
    for (int off = 16; off; off >>= 1) {
        d1 += __shfl_xor(d1, off, 32);
        d2 += __shfl_xor(d2, off, 32);
    }
    if (lane == 0) { s1[row] = d1; s2[row] = d2; }
}

// ---------------------------------------------------------------------------
// Pass 1: global max of leaky_relu(s1[src]+s2[tgt]) over all edges
// ---------------------------------------------------------------------------
__global__ __launch_bounds__(256) void heg_edge_max(
    const int* __restrict__ src, const int* __restrict__ tgt,
    const float* __restrict__ s1, const float* __restrict__ s2,
    unsigned* __restrict__ maxkey, int E)
{
    int i = blockIdx.x * blockDim.x + threadIdx.x;
    float l = -INFINITY;
    if (i < E) {
        float v = s1[src[i]] + s2[tgt[i]];
        l = (v > 0.f) ? v : 0.2f * v;
    }
    #pragma unroll
    for (int off = 16; off; off >>= 1) l = fmaxf(l, __shfl_xor(l, off, 32));
    __shared__ float red[8];
    if ((threadIdx.x & 31) == 0) red[threadIdx.x >> 5] = l;
    __syncthreads();
    if (threadIdx.x < 8) {
        float v = red[threadIdx.x];
        #pragma unroll
        for (int off = 4; off; off >>= 1) v = fmaxf(v, __shfl_xor(v, off, 32));
        if (threadIdx.x == 0) atomicMax(maxkey, fkey(v));
    }
}

// ---------------------------------------------------------------------------
// Pass 2: global sum of exp(l - max)
// ---------------------------------------------------------------------------
__global__ __launch_bounds__(256) void heg_edge_sum(
    const int* __restrict__ src, const int* __restrict__ tgt,
    const float* __restrict__ s1, const float* __restrict__ s2,
    const unsigned* __restrict__ maxkey, float* __restrict__ sumptr, int E)
{
    float mx = fval(*maxkey);
    int i = blockIdx.x * blockDim.x + threadIdx.x;
    float s = 0.f;
    if (i < E) {
        float v = s1[src[i]] + s2[tgt[i]];
        float l = (v > 0.f) ? v : 0.2f * v;
        s = expf(l - mx);
    }
    #pragma unroll
    for (int off = 16; off; off >>= 1) s += __shfl_xor(s, off, 32);
    __shared__ float red[8];
    if ((threadIdx.x & 31) == 0) red[threadIdx.x >> 5] = s;
    __syncthreads();
    if (threadIdx.x < 8) {
        float v = red[threadIdx.x];
        #pragma unroll
        for (int off = 4; off; off >>= 1) v += __shfl_xor(v, off, 32);
        if (threadIdx.x == 0) atomicAdd(sumptr, v);
    }
}

// ---------------------------------------------------------------------------
// Scatter: Xout[tgt, colofs + c] += alpha(e) * Wh[src, c]   (H/4 threads/edge)
// ---------------------------------------------------------------------------
__global__ __launch_bounds__(256) void heg_edge_scatter(
    const int* __restrict__ src, const int* __restrict__ tgt,
    const float* __restrict__ s1, const float* __restrict__ s2,
    const float* __restrict__ Wh, float* __restrict__ Xout,
    const unsigned* __restrict__ maxkey, const float* __restrict__ sumptr,
    int E, int H, int ldx, int colofs)
{
    int tpe = H >> 2;
    long long gid = (long long)blockIdx.x * blockDim.x + threadIdx.x;
    long long e = gid / tpe;
    int c4 = (int)(gid % tpe) << 2;
    if (e >= E) return;
    int s = src[e], t = tgt[e];
    float v = s1[s] + s2[t];
    float l = (v > 0.f) ? v : 0.2f * v;
    float alpha = expf(l - fval(*maxkey)) / (*sumptr);
    const float4 w = *(const float4*)&Wh[(size_t)s * H + c4];
    float* dst = &Xout[(size_t)t * ldx + colofs + c4];
    atomicAdd(dst + 0, alpha * w.x);
    atomicAdd(dst + 1, alpha * w.y);
    atomicAdd(dst + 2, alpha * w.z);
    atomicAdd(dst + 3, alpha * w.w);
}

// ---------------------------------------------------------------------------
// LayerNorm + ELU, wave per row (H <= 256, H % 32 == 0)
// ---------------------------------------------------------------------------
__global__ __launch_bounds__(256) void heg_ln_elu(
    const float* __restrict__ X, const float* __restrict__ g,
    const float* __restrict__ b, float* __restrict__ Y, int N, int H)
{
    int row  = blockIdx.x * (blockDim.x >> 5) + (threadIdx.x >> 5);
    int lane = threadIdx.x & 31;
    if (row >= N) return;
    int per = H >> 5;
    float vals[8];
    float s = 0.f;
    for (int i = 0; i < per; ++i) {
        vals[i] = X[(size_t)row * H + lane + (i << 5)];
        s += vals[i];
    }
    #pragma unroll
    for (int off = 16; off; off >>= 1) s += __shfl_xor(s, off, 32);
    float mean = s / (float)H;
    float vs = 0.f;
    for (int i = 0; i < per; ++i) { float d = vals[i] - mean; vs += d * d; }
    #pragma unroll
    for (int off = 16; off; off >>= 1) vs += __shfl_xor(vs, off, 32);
    float inv = rsqrtf(vs / (float)H + 1e-5f);
    for (int i = 0; i < per; ++i) {
        int c = lane + (i << 5);
        float v = (vals[i] - mean) * inv * g[c] + b[c];
        Y[(size_t)row * H + c] = (v > 0.f) ? v : (expf(v) - 1.f);
    }
}

// ---------------------------------------------------------------------------
// evidence head: ev = softplus(Hh @ w2 + b2) + 1 ; optional accumulate
// ---------------------------------------------------------------------------
__global__ __launch_bounds__(256) void heg_ev_head(
    const float* __restrict__ Hh, const float* __restrict__ w2,
    const float* __restrict__ b2, float* __restrict__ evout,
    int N, int Hc, int accumulate)
{
    int i = blockIdx.x * blockDim.x + threadIdx.x;
    if (i >= N) return;
    float acc0 = b2[0], acc1 = b2[1], acc2 = b2[2];
    for (int c = 0; c < Hc; ++c) {
        float h = Hh[(size_t)i * Hc + c];
        acc0 += h * w2[c * 3 + 0];
        acc1 += h * w2[c * 3 + 1];
        acc2 += h * w2[c * 3 + 2];
    }
    float e0 = softplus_f(acc0) + 1.f;
    float e1 = softplus_f(acc1) + 1.f;
    float e2 = softplus_f(acc2) + 1.f;
    if (accumulate) {
        evout[(size_t)i * 3 + 0] += e0;
        evout[(size_t)i * 3 + 1] += e1;
        evout[(size_t)i * 3 + 2] += e2;
    } else {
        evout[(size_t)i * 3 + 0] = e0;
        evout[(size_t)i * 3 + 1] = e1;
        evout[(size_t)i * 3 + 2] = e2;
    }
}

// ---------------------------------------------------------------------------
// Final: x_out = elu(LN(x2)) + res ; ev_out = (ev1sum/4 + ev2)/2   (H = 128)
// ---------------------------------------------------------------------------
__global__ __launch_bounds__(256) void heg_final(
    const float* __restrict__ x2, const float* __restrict__ g,
    const float* __restrict__ b, const float* __restrict__ res,
    const float* __restrict__ ev1s, const float* __restrict__ ev2,
    float* __restrict__ out, int N, int H)
{
    int row  = blockIdx.x * (blockDim.x >> 5) + (threadIdx.x >> 5);
    int lane = threadIdx.x & 31;
    if (row >= N) return;
    float vals[4];
    float s = 0.f;
    #pragma unroll
    for (int i = 0; i < 4; ++i) {
        vals[i] = x2[(size_t)row * H + lane + (i << 5)];
        s += vals[i];
    }
    #pragma unroll
    for (int off = 16; off; off >>= 1) s += __shfl_xor(s, off, 32);
    float mean = s / (float)H;
    float vs = 0.f;
    #pragma unroll
    for (int i = 0; i < 4; ++i) { float d = vals[i] - mean; vs += d * d; }
    #pragma unroll
    for (int off = 16; off; off >>= 1) vs += __shfl_xor(vs, off, 32);
    float inv = rsqrtf(vs / (float)H + 1e-5f);
    #pragma unroll
    for (int i = 0; i < 4; ++i) {
        int c = lane + (i << 5);
        float v = (vals[i] - mean) * inv * g[c] + b[c];
        v = (v > 0.f) ? v : (expf(v) - 1.f);
        out[(size_t)row * H + c] = v + res[(size_t)row * H + c];
    }
    if (lane < 3) {
        float ev = (ev1s[(size_t)row * 3 + lane] * 0.25f +
                    ev2[(size_t)row * 3 + lane]) * 0.5f;
        out[(size_t)N * H + (size_t)row * 3 + lane] = ev;
    }
}

// ---------------------------------------------------------------------------
// Host orchestration
// ---------------------------------------------------------------------------
extern "C" void kernel_launch(void* const* d_in, const int* in_sizes, int n_in,
                              void* d_out, int out_size, void* d_ws, size_t ws_size,
                              hipStream_t stream) {
    const int N = 30000, E = 480000, IN = 768, HID = 256, OUT = 128, NH = 4;

    const float* x        = (const float*)d_in[0];
    const int*   eidx     = (const int*)d_in[1];
    const float* W_heads  = (const float*)d_in[2];
    const float* a_heads  = (const float*)d_in[3];
    const float* ev1w_h   = (const float*)d_in[4];
    const float* ev1b_h   = (const float*)d_in[5];
    const float* ev2w_h   = (const float*)d_in[6];
    const float* ev2b_h   = (const float*)d_in[7];
    const float* agg_w    = (const float*)d_in[8];
    const float* agg_b    = (const float*)d_in[9];
    const float* ln1_g    = (const float*)d_in[10];
    const float* ln1_b    = (const float*)d_in[11];
    const float* W2       = (const float*)d_in[12];
    const float* a2       = (const float*)d_in[13];
    const float* ev1w2    = (const float*)d_in[14];
    const float* ev1b2    = (const float*)d_in[15];
    const float* ev2w2    = (const float*)d_in[16];
    const float* ev2b2    = (const float*)d_in[17];
    const float* ln2_g    = (const float*)d_in[18];
    const float* ln2_b    = (const float*)d_in[19];
    const float* res_w    = (const float*)d_in[20];
    const float* res_b    = (const float*)d_in[21];

    const int* src = eidx;
    const int* tgt = eidx + E;

    // workspace layout (floats)
    float* ws    = (float*)d_ws;
    float* x_cat = ws;                                   // N*1024
    float* bufA  = x_cat + (size_t)N * (NH * HID);       // N*256
    float* h1    = bufA  + (size_t)N * HID;              // N*256
    float* bufB  = h1    + (size_t)N * HID;              // N*128
    float* x2    = bufB  + (size_t)N * OUT;              // N*128
    float* ev1s  = x2    + (size_t)N * OUT;              // N*3
    float* ev2   = ev1s  + (size_t)N * 3;                // N*3
    float* s1    = ev2   + (size_t)N * 3;                // N
    float* s2    = s1    + (size_t)N;                    // N
    unsigned* stats = (unsigned*)(s2 + (size_t)N);       // [0]=maxkey [1]=sum
    float* sumptr   = (float*)stats + 1;

    auto gemm = [&](const float* A, const float* B, float* C, const float* bias,
                    int M, int Nn, int K, int lda, int ldb, int ldc, int relu) {
        dim3 grid((Nn + GBN - 1) / GBN, (M + GBM - 1) / GBM);
        heg_gemm_bf16<<<grid, 256, 0, stream>>>(A, B, C, bias, M, Nn, K, lda, ldb, ldc, relu);
    };

    const int rowBlocks = (N + 7) / 8;             // wave-per-row kernels
    const int edgeBlocks = (E + 255) / 256;

    hipMemsetAsync(x_cat, 0, (size_t)N * NH * HID * sizeof(float), stream);
    hipMemsetAsync(ev1s, 0, (size_t)N * 3 * sizeof(float), stream);

    // ---- layer 1: per head ----
    for (int h = 0; h < NH; ++h) {
        // Wh = x @ W_h  -> bufA [N,256]
        gemm(x, W_heads + (size_t)h * IN * HID, bufA, nullptr,
             N, HID, IN, IN, HID, HID, 0);
        // s1/s2
        heg_rowdot2<<<rowBlocks, 256, 0, stream>>>(
            bufA, a_heads + (size_t)h * 2 * HID, s1, s2, N, HID);
        // global softmax stats
        hipMemsetAsync(stats, 0, 2 * sizeof(unsigned), stream);
        heg_edge_max<<<edgeBlocks, 256, 0, stream>>>(src, tgt, s1, s2, stats, E);
        heg_edge_sum<<<edgeBlocks, 256, 0, stream>>>(src, tgt, s1, s2, stats, sumptr, E);
        // scatter into x_cat columns [h*256, h*256+256)
        {
            long long work = (long long)E * (HID / 4);
            int blocks = (int)((work + 255) / 256);
            heg_edge_scatter<<<blocks, 256, 0, stream>>>(
                src, tgt, s1, s2, bufA, x_cat, stats, sumptr,
                E, HID, NH * HID, h * HID);
        }
        // evidence: hh = relu(x_new @ ev1w + b) -> bufB [N,128]
        gemm(x_cat + h * HID, ev1w_h + (size_t)h * HID * (HID / 2), bufB,
             ev1b_h + (size_t)h * (HID / 2),
             N, HID / 2, HID, NH * HID, HID / 2, HID / 2, 1);
        // ev1s += softplus(hh @ ev2w + b) + 1
        heg_ev_head<<<(N + 255) / 256, 256, 0, stream>>>(
            bufB, ev2w_h + (size_t)h * (HID / 2) * 3, ev2b_h + (size_t)h * 3,
            ev1s, N, HID / 2, 1);
    }

    // ---- aggregation ----
    gemm(x_cat, agg_w, bufA, agg_b, N, HID, NH * HID, NH * HID, HID, HID, 0);
    heg_ln_elu<<<rowBlocks, 256, 0, stream>>>(bufA, ln1_g, ln1_b, h1, N, HID);

    // ---- layer 2 ----
    gemm(h1, W2, bufB, nullptr, N, OUT, HID, HID, OUT, OUT, 0);       // Wh2
    heg_rowdot2<<<rowBlocks, 256, 0, stream>>>(bufB, a2, s1, s2, N, OUT);
    hipMemsetAsync(stats, 0, 2 * sizeof(unsigned), stream);
    heg_edge_max<<<edgeBlocks, 256, 0, stream>>>(src, tgt, s1, s2, stats, E);
    heg_edge_sum<<<edgeBlocks, 256, 0, stream>>>(src, tgt, s1, s2, stats, sumptr, E);
    hipMemsetAsync(x2, 0, (size_t)N * OUT * sizeof(float), stream);
    {
        long long work = (long long)E * (OUT / 4);
        int blocks = (int)((work + 255) / 256);
        heg_edge_scatter<<<blocks, 256, 0, stream>>>(
            src, tgt, s1, s2, bufB, x2, stats, sumptr, E, OUT, OUT, 0);
    }
    // evidence2: hh2 = relu(x2 @ ev1w2 + b) -> bufA [N,64]
    gemm(x2, ev1w2, bufA, ev1b2, N, OUT / 2, OUT, OUT, OUT / 2, OUT / 2, 1);
    heg_ev_head<<<(N + 255) / 256, 256, 0, stream>>>(
        bufA, ev2w2, ev2b2, ev2, N, OUT / 2, 0);
    // residual: res = h1 @ res_w + res_b -> bufB (Wh2 no longer needed)
    gemm(h1, res_w, bufB, res_b, N, OUT, HID, HID, OUT, OUT, 0);
    // final fuse
    heg_final<<<rowBlocks, 256, 0, stream>>>(
        x2, ln2_g, ln2_b, bufB, ev1s, ev2, (float*)d_out, N, OUT);

    (void)in_sizes; (void)n_in; (void)out_size; (void)ws_size;
}
